// CorrTorch_47871705481364
// MI455X (gfx1250) — matrix operand order repaired and checked by reading, not compiled, not run on hardware
//
#include <hip/hip_runtime.h>

typedef __attribute__((ext_vector_type(16))) __bf16 v16bf;
typedef __attribute__((ext_vector_type(8)))  float  v8f;
typedef __attribute__((ext_vector_type(8)))  unsigned int v8u;

namespace {
constexpr int Bn = 8, C = 256, H = 96, W = 192;
constexpr int TW = 16;               // WMMA tile (M = N = 16 pixels)
constexpr int ND = 5;                // displacements per axis
constexpr int KC = 32;               // K per bf16 WMMA
constexpr int NK = C / KC;           // 8 K-chunks
constexpr int sC = H * W;            // channel stride (elements)
constexpr long long sB = (long long)C * sC;
constexpr int tilesPerRow = W / TW;          // 12
constexpr int tilesPerImg = tilesPerRow * H; // 1152
constexpr int WAVES = 8;                     // waves per block
constexpr long long oSB = 25LL * sC;         // output batch stride
}

__global__ __launch_bounds__(WAVES * 32)
void corr25_wmma_bf16(const float* __restrict__ in1,
                      const float* __restrict__ in2,
                      float* __restrict__ out)
{
    __shared__ float lds[WAVES][2][8][32];

    const int lane = threadIdx.x & 31;
    const int wv   = threadIdx.x >> 5;
    // Wave-uniform tile id; readfirstlane makes uniformity visible to the
    // compiler so row-bound checks lower to scalar branches (no exec churn).
    const int tile = __builtin_amdgcn_readfirstlane((int)(blockIdx.x * WAVES + wv));

    const int b  = tile / tilesPerImg;
    const int r0 = tile % tilesPerImg;
    const int h  = r0 / tilesPerRow;
    const int w0 = (r0 % tilesPerRow) * TW;

    const int  mrow = lane & 15;      // M (A) / N (B) index within half-wave
    const bool hi   = lane >= 16;

    // A base: in1[b, :, h, w0 + mrow], channel-strided (always in bounds)
    const float* a_base = in1 + (long long)b * sB + (long long)h * W + (w0 + mrow);
    const float* b_img  = in2 + (long long)b * sB;

    // B-tile column geometry: fixed per (lane, t) -> hoist out of all loops.
    // Loads always use a clamped (valid) address; OOB columns are zeroed by
    // AND-ing the packed bf16 fragment with bmask (branchless, no exec churn).
    int      wbc[2];
    unsigned bmask[2];
#pragma unroll
    for (int t = 0; t < 2; ++t) {
        const int wb = w0 - 4 + t * 16 + mrow;
        int c = wb < 0 ? 0 : wb;
        wbc[t]   = c >= W ? (W - 1) : c;
        bmask[t] = (wb >= 0 && wb < W) ? 0xFFFFFFFFu : 0u;
    }

    v8f acc[ND][2];
#pragma unroll
    for (int dyi = 0; dyi < ND; ++dyi)
#pragma unroll
        for (int t = 0; t < 2; ++t)
            acc[dyi][t] = (v8f){0.f, 0.f, 0.f, 0.f, 0.f, 0.f, 0.f, 0.f};

#pragma unroll 1
    for (int kc = 0; kc < NK; ++kc) {
        const int c0 = kc * KC;

        // Prefetch next A chunk into near caches (speculative; dropped if OOB)
        __builtin_prefetch(a_base + (long long)(c0 + KC) * sC, 0, 3);

        // ---- A fragment: 16(M px) x 32(K ch) bf16, ISA VGPR layout ----
        // lanes 0-15: v<4 -> K=2v,2v+1 ; v>=4 -> K=2v+8,2v+9 ; lanes 16-31: +8
        v16bf afrag;
#pragma unroll
        for (int v = 0; v < 8; ++v) {
            const int k = ((v < 4) ? 2 * v : 2 * v + 8) + (hi ? 8 : 0);
            const float* p = a_base + (long long)(c0 + k) * sC;
            afrag[2 * v]     = (__bf16)p[0];
            afrag[2 * v + 1] = (__bf16)p[sC];
        }

#pragma unroll
        for (int dyi = 0; dyi < ND; ++dyi) {
            const int hy = h + 2 * dyi - 4;           // uniform
            if (hy < 0 || hy >= H) continue;          // scalar branch; pad = 0
            const float* row2 = b_img + (long long)hy * W;

#pragma unroll
            for (int t = 0; t < 2; ++t) {
                // B fragment: 32(K) x 16(N px); N = column = in2 pixel
                // lanes 0-15: v<4 -> K=2v,2v+1 ; v>=4 -> K=2v+8,2v+9 ; lanes 16-31: +8
                const int wcol = wbc[t];
                v16bf bfrag;
#pragma unroll
                for (int v = 0; v < 8; ++v) {
                    const int k = ((v < 4) ? 2 * v : 2 * v + 8) + (hi ? 8 : 0);
                    bfrag[2 * v]     = (__bf16)row2[(long long)(c0 + k)     * sC + wcol];
                    bfrag[2 * v + 1] = (__bf16)row2[(long long)(c0 + k + 1) * sC + wcol];
                }
                // Zero out-of-range columns (packed AND, branchless)
                v8u bu = __builtin_bit_cast(v8u, bfrag);
                bu &= bmask[t];
                bfrag = __builtin_bit_cast(v16bf, bu);

                acc[dyi][t] = __builtin_amdgcn_wmma_f32_16x16x32_bf16(
                    false, afrag, false, bfrag, (short)0, acc[dyi][t],
                    false, false);
            }
        }
    }

    // ---- Diagonal extraction via per-wave LDS; /C and store ----
    float* outb = out + (long long)b * oSB + (long long)h * W + w0;
#pragma unroll
    for (int dyi = 0; dyi < ND; ++dyi) {
#pragma unroll
        for (int t = 0; t < 2; ++t)
#pragma unroll
            for (int r = 0; r < 8; ++r)
                lds[wv][t][r][lane] = acc[dyi][t][r];
        // LDS ops are in-order within a wave; wait for data + block reordering
        asm volatile("s_wait_dscnt 0" ::: "memory");

        if (lane < 16) {
            const int m = lane;
#pragma unroll
            for (int dxi = 0; dxi < ND; ++dxi) {
                const int n0 = m + 2 * dxi;        // 0..23
                const int t  = n0 >> 4;            // which B-tile
                const int n  = n0 & 15;
                const int r  = m & 7;              // C/D layout: VGPR row
                const int sl = n + ((m >= 8) ? 16 : 0);
                const float val = lds[wv][t][r][sl] * (1.0f / 256.0f);
                // Output is streamed once; keep it out of L2 (in2 reuse wins)
                __builtin_nontemporal_store(
                    val, outb + (long long)(dyi * ND + dxi) * sC + m);
            }
        }
        asm volatile("" ::: "memory");  // keep next dyi's LDS stores after reads
    }
}

extern "C" void kernel_launch(void* const* d_in, const int* in_sizes, int n_in,
                              void* d_out, int out_size, void* d_ws, size_t ws_size,
                              hipStream_t stream) {
    const float* in1 = (const float*)d_in[0];
    const float* in2 = (const float*)d_in[1];
    float* out = (float*)d_out;

    const int totalTiles = Bn * tilesPerImg;     // 9216 waves of work
    const int blocks     = totalTiles / WAVES;   // 1152 (exact)
    corr25_wmma_bf16<<<blocks, WAVES * 32, 0, stream>>>(in1, in2, out);
}
